// UnifiedQLSTM_65481071400013
// MI455X (gfx1250) — matrix operand order repaired and checked by reading, not compiled, not run on hardware
//
#include <hip/hip_runtime.h>
#include <hip/hip_bf16.h>

#define T_STEPS 512
#define BATCH   64
#define DDIM    1024
#define HDIM    1024
#define KDIM    2048   // D + H
#define NGATES  4
#define GSTRIDE 65     // LDS row stride (floats) for gate buffer

#define KC      128                 // K-stage size staged into LDS
#define NSTAGE  (KDIM / KC)         // 16
#define APAD    (KC + 8)            // padded LDS row stride (bf16 elems)
#define BUFELEM (BATCH * APAD)      // 8704 bf16 per A buffer
#define BUFBYTE (BUFELEM * 2)       // 17408 bytes

typedef __attribute__((ext_vector_type(16))) __bf16 v16bf;
typedef __attribute__((ext_vector_type(8)))  float  v8f;

union BFVec { v16bf v; uint4 q[2]; };

// ---------------------------------------------------------------------------
// Prologue: fp32 -> bf16 conversion of the input sequence [T, B, D]
// ---------------------------------------------------------------------------
__global__ __launch_bounds__(256) void qlstm_convert_x(
    const float* __restrict__ x, __bf16* __restrict__ xb, int n) {
    int i = blockIdx.x * 256 + threadIdx.x;
    if (i < n) xb[i] = (__bf16)x[i];
}

// ---------------------------------------------------------------------------
// Prologue: transpose+convert the four gate weights into N-major bf16:
//   Wt[g*HDIM + j][k] = Wg[k][j]
// ---------------------------------------------------------------------------
__global__ __launch_bounds__(256) void qlstm_convert_w(
    const float* __restrict__ Wf, const float* __restrict__ Wi,
    const float* __restrict__ Wg, const float* __restrict__ Wo,
    __bf16* __restrict__ Wt) {
    int idx = blockIdx.x * 256 + threadIdx.x;       // over 4*1024*2048
    if (idx >= NGATES * HDIM * KDIM) return;
    int g   = idx >> 21;                            // HDIM*KDIM = 2^21
    int rem = idx & ((1 << 21) - 1);
    int j   = rem >> 11;                            // KDIM = 2^11
    int k   = rem & (KDIM - 1);
    const float* W = (g == 0) ? Wf : (g == 1) ? Wi : (g == 2) ? Wg : Wo;
    Wt[idx] = (__bf16)W[(size_t)k * HDIM + j];
}

// ---------------------------------------------------------------------------
// Prologue: zero initial h (bf16 buffer 0) and c (fp32)
// ---------------------------------------------------------------------------
__global__ __launch_bounds__(256) void qlstm_init_state(
    __bf16* __restrict__ Hb0, float* __restrict__ C) {
    int i = blockIdx.x * 256 + threadIdx.x;
    if (i < BATCH * HDIM) {
        Hb0[i] = (__bf16)0.0f;
        C[i]   = 0.0f;
    }
}

// ---------------------------------------------------------------------------
// Async-copy one K-stage of the activation matrix A[64][KC] into LDS.
// A columns [kb, kb+KC) come from X (kb < D) or from h_{t-1} (kb >= D).
// Each of the 128 threads issues 8 global_load_async_to_lds_b128 ops.
// ---------------------------------------------------------------------------
__device__ __forceinline__ void qlstm_issue_stage(
    const __bf16* __restrict__ xbase, const __bf16* __restrict__ hbase,
    int kb, unsigned int lds_off, int tid) {
    const __bf16* src = (kb < DDIM) ? xbase : hbase;
    const int kloc    = (kb < DDIM) ? kb : (kb - DDIM);
    #pragma unroll
    for (int it = 0; it < 8; ++it) {
        int c   = tid + it * 128;          // b128 chunk id, 0..1023
        int r   = c >> 4;                  // row (16 chunks per row)
        int col = (c & 15) * 8;            // element column within stage
        unsigned long long g =
            (unsigned long long)(src + (size_t)r * 1024 + kloc + col);
        unsigned int l = lds_off + (unsigned int)(r * APAD + col) * 2u;
        asm volatile("global_load_async_to_lds_b128 %0, %1, off"
                     :: "v"(l), "v"(g) : "memory");
    }
}

// ---------------------------------------------------------------------------
// One LSTM time step.
//  grid  = 64 blocks (hidden slices of 16 columns)
//  block = 128 threads = 4 waves; wave w owns M-tile w (batch rows 16w..16w+15)
//  and accumulates all four gates for the slice:
//          M=16 x N=16x4gates x K=2048 via v_wmma_f32_16x16x32_bf16.
//  One A fragment feeds 4 WMMAs (one per gate); A (LDS) and B (global)
//  are double-buffered across K-chunks; stages are double-buffered async.
// ---------------------------------------------------------------------------
__global__ __launch_bounds__(128) void qlstm_step(
    int t,
    const __bf16* __restrict__ Xb,   // [T*B, D] bf16
    const __bf16* __restrict__ Wt,   // [4*H, K] bf16 (N-major)
    const __bf16* __restrict__ Hr,   // [B, H] bf16 (read, h_{t-1})
    __bf16*       __restrict__ Hw,   // [B, H] bf16 (write, h_t)
    float*        __restrict__ C,    // [B, H] fp32 cell state
    const float* __restrict__ bf_, const float* __restrict__ bi_,
    const float* __restrict__ bg_, const float* __restrict__ bo_,
    float* __restrict__ outT)        // [B, H] fp32 h_t output slab
{
    __shared__ alignas(16) __bf16 lds_a[2 * BUFELEM];       // 34,816 B
    __shared__ alignas(16) float  gbuf[NGATES * 16 * GSTRIDE];

    const int tid   = threadIdx.x;
    const int lane  = tid & 31;
    const int mt    = tid >> 5;                // wave id == M-tile (0..3)
    const int j0    = blockIdx.x * 16;         // hidden-column slice
    const int row   = lane & 15;               // A row / B column within tile
    const int khalf = lane >> 4;               // which K half this lane holds
    const int koffA = khalf * 8;               // A: K{0-7,16-23} / {8-15,24-31}

    v8f acc[NGATES] = {};                      // one accumulator per gate

    // Per-gate weight row pointers (B operand: lane = column, khalf: K{0-7,16-23} / {8-15,24-31}).
    const __bf16* Bg[NGATES];
    #pragma unroll
    for (int g = 0; g < NGATES; ++g)
        Bg[g] = Wt + ((size_t)(g * HDIM + j0 + row)) * KDIM + khalf * 8;

    const __bf16* Abase = Xb + ((size_t)t * BATCH) * DDIM;
    const unsigned int lbase = (unsigned int)(size_t)(&lds_a[0]);

    // Kick off stage 0 into buffer 0.
    qlstm_issue_stage(Abase, Hr, 0, lbase, tid);

    for (int s = 0; s < NSTAGE; ++s) {
        const int kb  = s * KC;
        const int buf = s & 1;

        // Preload first K-chunk's B fragments (global; overlaps async wait).
        BFVec bbuf[2][NGATES];
        #pragma unroll
        for (int g = 0; g < NGATES; ++g) {
            const __bf16* bp = Bg[g] + kb;
            bbuf[0][g].q[0] = *(const uint4*)(bp);
            bbuf[0][g].q[1] = *(const uint4*)(bp + 16);
        }

        // Launch the next stage's async copy, then wait for *this* stage.
        if (s + 1 < NSTAGE) {
            qlstm_issue_stage(Abase, Hr, kb + KC,
                              lbase + (unsigned int)((buf ^ 1) * BUFBYTE), tid);
            asm volatile("s_wait_asynccnt 0x8" ::: "memory");
        } else {
            asm volatile("s_wait_asynccnt 0x0" ::: "memory");
        }
        __syncthreads();                       // stage data visible to all waves

        const __bf16* ab = &lds_a[buf * BUFELEM] + (mt * 16 + row) * APAD + koffA;

        // Preload first K-chunk's A fragment (LDS).
        BFVec abuf[2];
        abuf[0].q[0] = *(const uint4*)(ab);
        abuf[0].q[1] = *(const uint4*)(ab + 16);

        #pragma unroll
        for (int u = 0; u < KC / 32; ++u) {    // 4 K-chunks per stage
            const int cur = u & 1, nxt = cur ^ 1;
            if (u + 1 < KC / 32) {
                // Double-buffer: next chunk's B (global) and A (LDS).
                #pragma unroll
                for (int g = 0; g < NGATES; ++g) {
                    const __bf16* bp = Bg[g] + kb + (u + 1) * 32;
                    bbuf[nxt][g].q[0] = *(const uint4*)(bp);
                    bbuf[nxt][g].q[1] = *(const uint4*)(bp + 16);
                }
                const __bf16* ap = ab + (u + 1) * 32;
                abuf[nxt].q[0] = *(const uint4*)(ap);
                abuf[nxt].q[1] = *(const uint4*)(ap + 16);
            }
            // One A fragment feeds all four gate WMMAs.
            #pragma unroll
            for (int g = 0; g < NGATES; ++g)
                acc[g] = __builtin_amdgcn_wmma_f32_16x16x32_bf16(
                    false, abuf[cur].v, false, bbuf[cur][g].v, (short)0,
                    acc[g], false, false);
        }
        __syncthreads();   // all waves done reading buf before it is re-filled
    }

    // ---- spill pre-activations: gbuf[gate][jl][b] -------------------------
    // C/D layout: VGPR r, lanes 0-15 -> M=r, lanes 16-31 -> M=r+8; N=lane&15
    {
        const int jl = lane & 15;
        #pragma unroll
        for (int g = 0; g < NGATES; ++g) {
            #pragma unroll
            for (int r = 0; r < 8; ++r) {
                int brow = mt * 16 + r + 8 * khalf;
                gbuf[(g * 16 + jl) * GSTRIDE + brow] = acc[g][r];
            }
        }
    }
    __syncthreads();

    // ---- LSTM cell update --------------------------------------------------
    for (int idx = tid; idx < 16 * BATCH; idx += 128) {
        int jl = idx >> 6;
        int b  = idx & (BATCH - 1);
        int j  = j0 + jl;
        float f = gbuf[(0 * 16 + jl) * GSTRIDE + b] + bf_[j];
        float i = gbuf[(1 * 16 + jl) * GSTRIDE + b] + bi_[j];
        float g = gbuf[(2 * 16 + jl) * GSTRIDE + b] + bg_[j];
        float o = gbuf[(3 * 16 + jl) * GSTRIDE + b] + bo_[j];
        f = 1.0f / (1.0f + expf(-f));
        i = 1.0f / (1.0f + expf(-i));
        g = tanhf(g);
        o = 1.0f / (1.0f + expf(-o));
        float c = f * C[(size_t)b * HDIM + j] + i * g;
        C[(size_t)b * HDIM + j] = c;
        float h = o * tanhf(c);
        outT[(size_t)b * HDIM + j] = h;
        Hw[(size_t)b * HDIM + j]   = (__bf16)h;
    }
}

// ---------------------------------------------------------------------------
// Epilogue: hx = outputs[T-1], cx = C
// ---------------------------------------------------------------------------
__global__ __launch_bounds__(256) void qlstm_finalize(
    const float* __restrict__ lastH, const float* __restrict__ C,
    float* __restrict__ hx, float* __restrict__ cx) {
    int i = blockIdx.x * 256 + threadIdx.x;
    if (i < BATCH * HDIM) {
        hx[i] = lastH[i];
        cx[i] = C[i];
    }
}

// ---------------------------------------------------------------------------
extern "C" void kernel_launch(void* const* d_in, const int* in_sizes, int n_in,
                              void* d_out, int out_size, void* d_ws, size_t ws_size,
                              hipStream_t stream) {
    const float* inp = (const float*)d_in[0];
    const float* Wf  = (const float*)d_in[1];
    const float* bf_ = (const float*)d_in[2];
    const float* Wi  = (const float*)d_in[3];
    const float* bi_ = (const float*)d_in[4];
    const float* Wg  = (const float*)d_in[5];
    const float* bg_ = (const float*)d_in[6];
    const float* Wo  = (const float*)d_in[7];
    const float* bo_ = (const float*)d_in[8];
    float* out = (float*)d_out;

    // Workspace layout (bytes):
    //   [0, 16 MiB)          Wt   bf16 [4*H][K]
    //   [16 MiB, 80 MiB)     Xb   bf16 [T*B][D]
    //   [80 MiB, +256 KiB)   Hb   bf16 [2][B][H]  (double buffer)
    //   next 256 KiB         C    fp32 [B][H]
    char* ws = (char*)d_ws;
    __bf16* Wt = (__bf16*)ws;
    __bf16* Xb = (__bf16*)(ws + ((size_t)16 << 20));
    __bf16* Hb = (__bf16*)(ws + ((size_t)80 << 20));
    float*  C  = (float*) (ws + ((size_t)80 << 20) + ((size_t)256 << 10));

    const int nx = T_STEPS * BATCH * DDIM;              // 33,554,432
    qlstm_convert_x<<<(nx + 255) / 256, 256, 0, stream>>>(inp, Xb, nx);

    const int nw = NGATES * HDIM * KDIM;                // 8,388,608
    qlstm_convert_w<<<(nw + 255) / 256, 256, 0, stream>>>(Wf, Wi, Wg, Wo, Wt);

    qlstm_init_state<<<(BATCH * HDIM) / 256, 256, 0, stream>>>(Hb, C);

    const size_t slab = (size_t)BATCH * HDIM;           // 65,536
    for (int t = 0; t < T_STEPS; ++t) {
        qlstm_step<<<HDIM / 16, 128, 0, stream>>>(
            t, Xb, Wt,
            Hb + (size_t)(t & 1) * slab,
            Hb + (size_t)((t + 1) & 1) * slab,
            C, bf_, bi_, bg_, bo_,
            out + (size_t)t * slab);
    }

    qlstm_finalize<<<(BATCH * HDIM + 255) / 256, 256, 0, stream>>>(
        out + (size_t)(T_STEPS - 1) * slab, C,
        out + (size_t)T_STEPS * slab,
        out + (size_t)T_STEPS * slab + slab);
}
